// GraphSageWithSampling_44744969290126
// MI455X (gfx1250) — matrix operand order repaired and checked by reading, not compiled, not run on hardware
//
#include <hip/hip_runtime.h>
#include <hip/hip_bf16.h>

// GraphSAGE-with-sampling forward for MI455X (gfx1250, wave32, WMMA).
// Pipeline: [convert weights->bf16] -> [h0 = gather+leaky(GEMM), stored bf16] ->
// [agg1 builds X1=[h0|mean(nbrs)] bf16] -> [fused dual-WMMA conv + norm -> h1 f32]
// -> [agg2 builds X2] -> [fused dual-WMMA conv -> out f32].
// h0 is kept in bf16: it only feeds bf16 GEMM operands and neighbor means, and
// this halves both the h0 store traffic and the random-row gather traffic.

#define S2N 8192
#define S1N (S2N * 9)   // 73728
#define S0N (S1N * 9)   // 663552

typedef __attribute__((ext_vector_type(16))) __bf16 v16bf;
typedef __attribute__((ext_vector_type(8)))  float  v8f;

union FragU { v16bf v; uint4 q[2]; };

__device__ __forceinline__ __bf16 f2bf(float x) {
  unsigned u = __float_as_uint(x);
  unsigned r = (u + 0x7FFFu + ((u >> 16) & 1u)) >> 16;   // round-to-nearest-even
  unsigned short s = (unsigned short)r;
  return __builtin_bit_cast(__bf16, s);
}

__device__ __forceinline__ float bf2f(__bf16 x) {
  unsigned u = ((unsigned)__builtin_bit_cast(unsigned short, x)) << 16;
  return __uint_as_float(u);
}

__device__ __forceinline__ float leakyf(float x) { return x > 0.f ? x : 0.1f * x; }

// A fragment (16x32 bf16, row-major source, ld = K stride).
// Lane l: row = row0 + (l&15); kbase = k0 + 8*(l>>4);
// elems 0..7 -> K = kbase+0..7 ; elems 8..15 -> K = kbase+16..23.
__device__ __forceinline__ v16bf load_a_bf16(const __bf16* __restrict__ A, int ld,
                                             int row0, int k0, int lane) {
  int r  = row0 + (lane & 15);
  int kb = k0 + ((lane >> 4) << 3);
  const __bf16* p = A + (size_t)r * ld + kb;
  FragU u;
  u.q[0] = *(const uint4*)(p);
  u.q[1] = *(const uint4*)(p + 16);
  return u.v;
}

// Same A fragment but from an f32 source, converted to bf16 in-register.
__device__ __forceinline__ v16bf load_a_f32(const float* __restrict__ A, int ld,
                                            int row0, int k0, int lane) {
  int r  = row0 + (lane & 15);
  int kb = k0 + ((lane >> 4) << 3);
  const float* p = A + (size_t)r * ld + kb;
  float4 a0 = *(const float4*)(p);
  float4 a1 = *(const float4*)(p + 4);
  float4 b0 = *(const float4*)(p + 16);
  float4 b1 = *(const float4*)(p + 20);
  v16bf f;
  f[0] = f2bf(a0.x);  f[1] = f2bf(a0.y);  f[2] = f2bf(a0.z);  f[3] = f2bf(a0.w);
  f[4] = f2bf(a1.x);  f[5] = f2bf(a1.y);  f[6] = f2bf(a1.z);  f[7] = f2bf(a1.w);
  f[8] = f2bf(b0.x);  f[9] = f2bf(b0.y);  f[10] = f2bf(b0.z); f[11] = f2bf(b0.w);
  f[12] = f2bf(b1.x); f[13] = f2bf(b1.y); f[14] = f2bf(b1.z); f[15] = f2bf(b1.w);
  return f;
}

// B fragment for B = W^T where W is (N x K) row-major bf16 (ldk = K).
// Lane l: n = n0 + (l&15); kbase = k0 + 8*(l>>4); elems 0..7 -> K = kbase+0..7 ; elems 8..15 -> K = kbase+16..23.
__device__ __forceinline__ v16bf load_b_wT(const __bf16* __restrict__ W, int ldk,
                                           int n0, int k0, int lane) {
  int n  = n0 + (lane & 15);
  int ks = k0 + ((lane >> 4) << 3);
  const __bf16* p = W + (size_t)n * ldk + ks;
  FragU u;
  u.q[0] = *(const uint4*)(p);
  u.q[1] = *(const uint4*)(p + 16);
  return u.v;
}

__device__ __forceinline__ v8f vzero8() {
  v8f z = {0.f, 0.f, 0.f, 0.f, 0.f, 0.f, 0.f, 0.f};
  return z;
}

// ---------------------------------------------------------------- weights -> bf16
__global__ __launch_bounds__(256) void convert_weights_kernel(
    const float* __restrict__ Wp, const float* __restrict__ W0,
    const float* __restrict__ Wagg0, const float* __restrict__ W1,
    const float* __restrict__ Wagg1, __bf16* __restrict__ out) {
  int i = blockIdx.x * blockDim.x + threadIdx.x;
  if (i < 32768)        out[i] = f2bf(Wp[i]);
  else if (i < 65536)   out[i] = f2bf(W0[i - 32768]);
  else if (i < 81920)   out[i] = f2bf(Wagg0[i - 65536]);
  else if (i < 114688)  out[i] = f2bf(W1[i - 81920]);
  else if (i < 131072)  out[i] = f2bf(Wagg1[i - 114688]);
}

// ------------------------------------------- h0 = emb[parent+1] + leaky(C @ Wp^T + bp)
// One wave = 16 rows x 128 cols (8 WMMA tiles). Block = 8 waves = 128 rows.
// Result stored bf16 to halve downstream traffic.
__global__ __launch_bounds__(256) void h0_gemm_kernel(
    const float* __restrict__ content0, const __bf16* __restrict__ WpB,
    const float* __restrict__ bp, const float* __restrict__ node_emb,
    const int* __restrict__ parent0, __bf16* __restrict__ h0) {
  int lane = threadIdx.x & 31;
  int wave = threadIdx.x >> 5;
  int row0 = blockIdx.x * 128 + wave * 16;

  v8f acc[8];
#pragma unroll
  for (int t = 0; t < 8; ++t) acc[t] = vzero8();

  for (int k0 = 0; k0 < 256; k0 += 32) {
    v16bf a = load_a_f32(content0, 256, row0, k0, lane);
#pragma unroll
    for (int t = 0; t < 8; ++t) {
      v16bf b = load_b_wT(WpB, 256, t * 16, k0, lane);
      acc[t] = __builtin_amdgcn_wmma_f32_16x16x32_bf16(
          false, a, false, b, (short)0, acc[t], false, false);
    }
  }

  int g = lane >> 4, nlo = lane & 15;
#pragma unroll
  for (int r = 0; r < 8; ++r) {
    int row = row0 + r + 8 * g;
    int par = parent0[row] + 1;
    const float* emb = node_emb + (size_t)par * 128;
    __bf16* orow = h0 + (size_t)row * 128;
#pragma unroll
    for (int t = 0; t < 8; ++t) {
      int col = t * 16 + nlo;
      orow[col] = f2bf(leakyf(acc[t][r] + bp[col]) + emb[col]);
    }
  }
}

// ------------------------------------------- layer-1 aggregation -> X1 (bf16, S1 x 256)
// Edge structure: self edge + 8 neighbors, h_dst == h0[i]  =>  h_agg = mean(8 nbrs).
__global__ __launch_bounds__(128) void agg1_kernel(
    const __bf16* __restrict__ h0, const int* __restrict__ src0,
    __bf16* __restrict__ X1) {
  int i = blockIdx.x;
  int c = threadIdx.x;
  const int* nbr = src0 + S1N + (size_t)i * 8;
  float s = 0.f;
#pragma unroll
  for (int j = 0; j < 8; ++j) s += bf2f(h0[(size_t)nbr[j] * 128 + c]);
  X1[(size_t)i * 256 + c]       = h0[(size_t)i * 128 + c];   // straight bf16 copy
  X1[(size_t)i * 256 + 128 + c] = f2bf(s * 0.125f);
}

// ------------------------------------------- layer-2 aggregation -> X2 (bf16, S2 x 256)
// h_dst = h0[:S2]; agg includes self edge h1[i]  =>  h_agg = (h1[i]+sum(nbr h1)-h0[i])/8.
__global__ __launch_bounds__(128) void agg2_kernel(
    const __bf16* __restrict__ h0, const float* __restrict__ h1,
    const int* __restrict__ src1, __bf16* __restrict__ X2) {
  int i = blockIdx.x;
  int c = threadIdx.x;
  const int* nbr = src1 + S2N + (size_t)i * 8;
  float s = h1[(size_t)i * 128 + c];
#pragma unroll
  for (int j = 0; j < 8; ++j) s += h1[(size_t)nbr[j] * 128 + c];
  float hd = bf2f(h0[(size_t)i * 128 + c]);
  X2[(size_t)i * 256 + c]       = h0[(size_t)i * 128 + c];   // straight bf16 copy
  X2[(size_t)i * 256 + 128 + c] = f2bf((s - hd) * 0.125f);
}

// ------------------------------------------- fused conv GEMM:
// out = act(X @ W^T + b) + act(X[:,128:] @ Wagg^T + bagg), optional row L2-norm.
// Both GEMMs share A fragments: Wagg accumulates only for k0 >= 128.
__global__ __launch_bounds__(256) void conv_gemm_kernel(
    const __bf16* __restrict__ X, const __bf16* __restrict__ W,
    const __bf16* __restrict__ Wagg, const float* __restrict__ b,
    const float* __restrict__ bagg, float* __restrict__ out, int last) {
  int lane = threadIdx.x & 31;
  int wave = threadIdx.x >> 5;
  int row0 = blockIdx.x * 128 + wave * 16;

  v8f acc1[8], acc2[8];
#pragma unroll
  for (int t = 0; t < 8; ++t) { acc1[t] = vzero8(); acc2[t] = vzero8(); }

  for (int k0 = 0; k0 < 256; k0 += 32) {
    v16bf a = load_a_bf16(X, 256, row0, k0, lane);
#pragma unroll
    for (int t = 0; t < 8; ++t) {
      v16bf bf = load_b_wT(W, 256, t * 16, k0, lane);
      acc1[t] = __builtin_amdgcn_wmma_f32_16x16x32_bf16(
          false, a, false, bf, (short)0, acc1[t], false, false);
    }
    if (k0 >= 128) {
#pragma unroll
      for (int t = 0; t < 8; ++t) {
        v16bf bf = load_b_wT(Wagg, 128, t * 16, k0 - 128, lane);
        acc2[t] = __builtin_amdgcn_wmma_f32_16x16x32_bf16(
            false, a, false, bf, (short)0, acc2[t], false, false);
      }
    }
  }

  int g = lane >> 4, nlo = lane & 15;
#pragma unroll
  for (int t = 0; t < 8; ++t) {
    int col = t * 16 + nlo;
    float bb = b[col], ba = bagg[col];
#pragma unroll
    for (int r = 0; r < 8; ++r) {
      float v1 = acc1[t][r] + bb;
      float v2 = acc2[t][r] + ba;
      if (!last) { v1 = leakyf(v1); v2 = leakyf(v2); }
      acc1[t][r] = v1 + v2;
    }
  }

  if (!last) {
    // Row L2 norm: each half-wave's 16 lanes jointly hold one 128-col row per r.
#pragma unroll
    for (int r = 0; r < 8; ++r) {
      float ss = 0.f;
#pragma unroll
      for (int t = 0; t < 8; ++t) ss += acc1[t][r] * acc1[t][r];
      ss += __shfl_xor(ss, 1, 32);
      ss += __shfl_xor(ss, 2, 32);
      ss += __shfl_xor(ss, 4, 32);
      ss += __shfl_xor(ss, 8, 32);
      float sc = 1.f / fmaxf(sqrtf(ss), 1e-6f);
#pragma unroll
      for (int t = 0; t < 8; ++t) acc1[t][r] *= sc;
    }
  }

#pragma unroll
  for (int r = 0; r < 8; ++r) {
    int row = row0 + r + 8 * g;
    float* orow = out + (size_t)row * 128;
#pragma unroll
    for (int t = 0; t < 8; ++t) orow[t * 16 + nlo] = acc1[t][r];
  }
}

// ---------------------------------------------------------------- launcher
extern "C" void kernel_launch(void* const* d_in, const int* in_sizes, int n_in,
                              void* d_out, int out_size, void* d_ws, size_t ws_size,
                              hipStream_t stream) {
  const float* node_emb = (const float*)d_in[0];
  const float* content0 = (const float*)d_in[1];
  const float* Wp    = (const float*)d_in[2];
  const float* bp    = (const float*)d_in[3];
  const float* W0    = (const float*)d_in[4];
  const float* b0    = (const float*)d_in[5];
  const float* Wagg0 = (const float*)d_in[6];
  const float* bagg0 = (const float*)d_in[7];
  const float* W1    = (const float*)d_in[8];
  const float* b1    = (const float*)d_in[9];
  const float* Wagg1 = (const float*)d_in[10];
  const float* bagg1 = (const float*)d_in[11];
  const int* parent0 = (const int*)d_in[12];
  const int* src0    = (const int*)d_in[13];
  const int* src1    = (const int*)d_in[15];

  // Workspace layout (bytes), all offsets 512B-aligned. Total ~250 MB.
  const size_t OFF_H0 = 0;
  const size_t OFF_X1 = OFF_H0 + (size_t)S0N * 128 * 2;   // h0 bf16
  const size_t OFF_H1 = OFF_X1 + (size_t)S1N * 256 * 2;   // X1 bf16
  const size_t OFF_X2 = OFF_H1 + (size_t)S1N * 128 * 4;   // h1 f32
  const size_t OFF_WB = OFF_X2 + (size_t)S2N * 256 * 2;   // X2 bf16

  char* ws = (char*)d_ws;
  __bf16* h0 = (__bf16*)(ws + OFF_H0);
  __bf16* X1 = (__bf16*)(ws + OFF_X1);
  float*  h1 = (float*)(ws + OFF_H1);
  __bf16* X2 = (__bf16*)(ws + OFF_X2);
  __bf16* wb = (__bf16*)(ws + OFF_WB);
  __bf16* WpB    = wb;
  __bf16* W0B    = wb + 32768;
  __bf16* Wagg0B = wb + 65536;
  __bf16* W1B    = wb + 81920;
  __bf16* Wagg1B = wb + 114688;

  convert_weights_kernel<<<512, 256, 0, stream>>>(Wp, W0, Wagg0, W1, Wagg1, wb);
  h0_gemm_kernel<<<S0N / 128, 256, 0, stream>>>(content0, WpB, bp, node_emb, parent0, h0);
  agg1_kernel<<<S1N, 128, 0, stream>>>(h0, src0, X1);
  conv_gemm_kernel<<<S1N / 128, 256, 0, stream>>>(X1, W0B, Wagg0B, b0, bagg0, h1, 0);
  agg2_kernel<<<S2N, 128, 0, stream>>>(h0, h1, src1, X2);
  conv_gemm_kernel<<<S2N / 128, 256, 0, stream>>>(X2, W1B, Wagg1B, b1, bagg1, (float*)d_out, 1);
}
